// MegNet_layer_50027779064052
// MI455X (gfx1250) — matrix operand order repaired and checked by reading, not compiled, not run on hardware
//
#include <hip/hip_runtime.h>
#include <stdint.h>

// ---------------------------------------------------------------------------
// MegNet layer for MI455X (gfx1250): bf16 WMMA (v_wmma_f32_16x16x32_bf16)
// for all GEMMs, f32 accumulation/atomics for segment reductions.
// ---------------------------------------------------------------------------

typedef __attribute__((ext_vector_type(16))) __bf16 v16bf;
typedef __attribute__((ext_vector_type(8)))  float  v8f;

#define SLOPE 0.01f
#define DD 128

__device__ __forceinline__ float lrelu(float v) { return v > 0.0f ? v : SLOPE * v; }

// WMMA wrapper: D = A(16x32 bf16) * B(32x16 bf16) + C(16x16 f32)
__device__ __forceinline__ v8f wmma_bf16(v16bf a, v16bf b, v8f c) {
    return __builtin_amdgcn_wmma_f32_16x16x32_bf16(false, a, false, b, (short)0, c,
                                                   false, false);
}

// A fragment (16x32 bf16; contiguous k order used by this code, not the ISA table order):
// lane L: row = L%16, half = L/16.
//   half 0: K = 0..15 contiguous; half 1: K = 16..31 (matches load_b).
__device__ __forceinline__ v16bf load_a_bf16(const __bf16* __restrict__ row,
                                             int half, int kbase) {
    v16bf a;
#pragma unroll
    for (int p = 0; p < 8; ++p) {
        int k = kbase + half * 16 + p * 2;
        a[2 * p]     = row[k];
        a[2 * p + 1] = row[k + 1];
    }
    return a;
}

__device__ __forceinline__ v16bf load_a_f32(const float* __restrict__ row,
                                            int half, int kbase, float scale) {
    v16bf a;
#pragma unroll
    for (int p = 0; p < 8; ++p) {
        int k = kbase + half * 16 + p * 2;
        a[2 * p]     = (__bf16)(row[k] * scale);
        a[2 * p + 1] = (__bf16)(row[k + 1] * scale);
    }
    return a;
}

// B fragment (16-bit B-matrix 32x16): lane L holds column L%16;
// lanes 0-15 hold K = kbase..kbase+15, lanes 16-31 hold kbase+16..kbase+31.
// Wt is stored transposed [128 out][K in] so this is 16 contiguous bf16.
__device__ __forceinline__ v16bf load_b(const __bf16* __restrict__ Wt, int K,
                                        int col, int half, int kbase) {
    return *(const v16bf*)(Wt + (size_t)col * K + kbase + half * 16);
}

// ---------------------------------------------------------------------------
// small utility kernels
// ---------------------------------------------------------------------------
__global__ void zero_kernel(uint32_t* p, size_t n) {
    size_t i = (size_t)blockIdx.x * blockDim.x + threadIdx.x;
    size_t stride = (size_t)gridDim.x * blockDim.x;
    for (; i < n; i += stride) p[i] = 0u;
}

// Wt[n*K + k] = bf16(W[k*128 + n])   (W is [K][128] row-major)
__global__ void transpose_w_kernel(const float* __restrict__ W,
                                   __bf16* __restrict__ Wt, int K) {
    int i = blockIdx.x * blockDim.x + threadIdx.x;
    if (i >= K * DD) return;
    int n = i & (DD - 1);
    int k = i >> 7;
    Wt[(size_t)n * K + k] = (__bf16)W[i];
}

// deg[dst]++, ne_cnt[g(dst)]++, nn_cnt[g(n)]++, graph_pool += graph_feats
__global__ void counts_kernel(const int* __restrict__ dst,
                              const int* __restrict__ n2g,
                              const float* __restrict__ graph_feats,
                              float* deg, float* nn_cnt, float* ne_cnt,
                              float* graph_pool, int N, int E) {
    long total = (long)N * DD;
    long stride = (long)gridDim.x * blockDim.x;
    for (long i = (long)blockIdx.x * blockDim.x + threadIdx.x; i < total; i += stride) {
        int n = (int)(i >> 7), c = (int)(i & (DD - 1));
        atomicAdd(graph_pool + (size_t)n2g[n] * DD + c, graph_feats[i]);
        if (i < E) {
            int d = dst[i];
            atomicAdd(deg + d, 1.0f);
            atomicAdd(ne_cnt + n2g[d], 1.0f);
        }
        if (i < N) atomicAdd(nn_cnt + n2g[i], 1.0f);
    }
}

// ---------------------------------------------------------------------------
// projection: out_bf = bf16(leaky_relu(X @ W + b)), X f32 [rows,128]
// one wave per 16-row tile, 8 col-tiles (full 128 output cols) per wave
// ---------------------------------------------------------------------------
__global__ __launch_bounds__(32) void proj_kernel(const float* __restrict__ X,
                                                  const __bf16* __restrict__ Wt,
                                                  const float* __restrict__ bias,
                                                  __bf16* __restrict__ out, int rows) {
    int lane = threadIdx.x;
    int r0 = blockIdx.x * 16;
    int mrow = lane & 15, half = lane >> 4;
    int r = r0 + mrow;
    if (r >= rows) r = rows - 1;
    const float* arow = X + (size_t)r * DD;

    v8f acc[8] = {};
#pragma unroll
    for (int ks = 0; ks < 4; ++ks) {
        v16bf a = load_a_f32(arow, half, ks * 32, 1.0f);
#pragma unroll
        for (int ct = 0; ct < 8; ++ct)
            acc[ct] = wmma_bf16(a, load_b(Wt, DD, ct * 16 + mrow, half, ks * 32), acc[ct]);
    }
#pragma unroll
    for (int ct = 0; ct < 8; ++ct) {
        int col = ct * 16 + mrow;
        float bv = bias[col];
#pragma unroll
        for (int g = 0; g < 8; ++g) {
            int er = r0 + g + (half << 3);
            if (er < rows)
                out[(size_t)er * DD + col] = (__bf16)lrelu(acc[ct][g] + bv);
        }
    }
}

// ---------------------------------------------------------------------------
// edge update: f_new = lrelu([h[src], h[dst], f, u[dst]] @ Weu + beu)
//   out_edge = f_new + edge_feats
//   hf_sum[dst]   += f_new   (f32 atomics)
//   edge_pool[g(dst)] += f_new
// ---------------------------------------------------------------------------
__global__ __launch_bounds__(32) void edge_update_kernel(
    const __bf16* __restrict__ h_bf, const __bf16* __restrict__ f_bf,
    const __bf16* __restrict__ u_bf, const __bf16* __restrict__ Weu_t,
    const float* __restrict__ beu, const int* __restrict__ src,
    const int* __restrict__ dst, const int* __restrict__ n2g,
    const float* __restrict__ edge_feats, float* __restrict__ out_edge,
    float* hf_sum, float* edge_pool, int E) {
    int lane = threadIdx.x;
    int r0 = blockIdx.x * 16;
    int mrow = lane & 15, half = lane >> 4;
    int e = r0 + mrow;
    if (e >= E) e = E - 1;
    int si = src[e], di = dst[e];
    const __bf16* arows[4] = {h_bf + (size_t)si * DD, h_bf + (size_t)di * DD,
                              f_bf + (size_t)e * DD,  u_bf + (size_t)di * DD};

    __builtin_prefetch(edge_feats + (size_t)e * DD, 0, 1);  // global_prefetch_b8

    v8f acc[8] = {};
#pragma unroll
    for (int s = 0; s < 4; ++s) {
        const __bf16* ar = arows[s];
#pragma unroll
        for (int ks = 0; ks < 4; ++ks) {
            v16bf a = load_a_bf16(ar, half, ks * 32);
#pragma unroll
            for (int ct = 0; ct < 8; ++ct)
                acc[ct] = wmma_bf16(
                    a, load_b(Weu_t, 4 * DD, ct * 16 + mrow, half, s * DD + ks * 32),
                    acc[ct]);
        }
    }

    int drow[8], grow[8];
#pragma unroll
    for (int g = 0; g < 8; ++g) {
        int er = r0 + g + (half << 3);
        int d = dst[er < E ? er : E - 1];
        drow[g] = d;
        grow[g] = n2g[d];
    }
#pragma unroll
    for (int ct = 0; ct < 8; ++ct) {
        int col = ct * 16 + mrow;
        float bv = beu[col];
#pragma unroll
        for (int g = 0; g < 8; ++g) {
            int er = r0 + g + (half << 3);
            if (er < E) {
                float v = lrelu(acc[ct][g] + bv);
                out_edge[(size_t)er * DD + col] = v + edge_feats[(size_t)er * DD + col];
                atomicAdd(hf_sum + (size_t)drow[g] * DD + col, v);
                atomicAdd(edge_pool + (size_t)grow[g] * DD + col, v);
            }
        }
    }
}

// ---------------------------------------------------------------------------
// node update: node_new = lrelu([h, hf_sum/deg, u] @ Wnu + bnu)
//   out_node = node_new + node_feats ; node_pool[g(n)] += node_new
// ---------------------------------------------------------------------------
__global__ __launch_bounds__(32) void node_update_kernel(
    const __bf16* __restrict__ h_bf, const float* __restrict__ hf_sum,
    const float* __restrict__ deg, const __bf16* __restrict__ u_bf,
    const __bf16* __restrict__ Wnu_t, const float* __restrict__ bnu,
    const int* __restrict__ n2g, const float* __restrict__ node_feats,
    float* __restrict__ out_node, float* node_pool, int N) {
    int lane = threadIdx.x;
    int r0 = blockIdx.x * 16;
    int mrow = lane & 15, half = lane >> 4;
    int r = r0 + mrow;
    if (r >= N) r = N - 1;
    const __bf16* a_h = h_bf + (size_t)r * DD;
    const float* a_hf = hf_sum + (size_t)r * DD;
    const __bf16* a_u = u_bf + (size_t)r * DD;
    float invd = 1.0f / fmaxf(deg[r], 1.0f);

    v8f acc[8] = {};
#pragma unroll
    for (int ks = 0; ks < 4; ++ks) {
        v16bf a = load_a_bf16(a_h, half, ks * 32);
#pragma unroll
        for (int ct = 0; ct < 8; ++ct)
            acc[ct] = wmma_bf16(a, load_b(Wnu_t, 3 * DD, ct * 16 + mrow, half, ks * 32),
                                acc[ct]);
    }
#pragma unroll
    for (int ks = 0; ks < 4; ++ks) {
        v16bf a = load_a_f32(a_hf, half, ks * 32, invd);
#pragma unroll
        for (int ct = 0; ct < 8; ++ct)
            acc[ct] = wmma_bf16(
                a, load_b(Wnu_t, 3 * DD, ct * 16 + mrow, half, DD + ks * 32), acc[ct]);
    }
#pragma unroll
    for (int ks = 0; ks < 4; ++ks) {
        v16bf a = load_a_bf16(a_u, half, ks * 32);
#pragma unroll
        for (int ct = 0; ct < 8; ++ct)
            acc[ct] = wmma_bf16(
                a, load_b(Wnu_t, 3 * DD, ct * 16 + mrow, half, 2 * DD + ks * 32), acc[ct]);
    }

    int gg[8];
#pragma unroll
    for (int g = 0; g < 8; ++g) {
        int nr = r0 + g + (half << 3);
        gg[g] = n2g[nr < N ? nr : N - 1];
    }
#pragma unroll
    for (int ct = 0; ct < 8; ++ct) {
        int col = ct * 16 + mrow;
        float bv = bnu[col];
#pragma unroll
        for (int g = 0; g < 8; ++g) {
            int nr = r0 + g + (half << 3);
            if (nr < N) {
                float v = lrelu(acc[ct][g] + bv);
                out_node[(size_t)nr * DD + col] = v + node_feats[(size_t)nr * DD + col];
                atomicAdd(node_pool + (size_t)gg[g] * DD + col, v);
            }
        }
    }
}

// pools_bf[b][0:128]=node_pool/nn, [128:256]=edge_pool/ne, [256:384]=graph_pool/nn
__global__ void pool_finalize_kernel(const float* node_pool, const float* edge_pool,
                                     const float* graph_pool, const float* nn_cnt,
                                     const float* ne_cnt, __bf16* pools_bf, int B) {
    int i = blockIdx.x * blockDim.x + threadIdx.x;
    if (i >= B * 3 * DD) return;
    int b = i / (3 * DD), j = i % (3 * DD);
    float v;
    if (j < DD)
        v = node_pool[b * DD + j] / fmaxf(nn_cnt[b], 1.0f);
    else if (j < 2 * DD)
        v = edge_pool[b * DD + (j - DD)] / fmaxf(ne_cnt[b], 1.0f);
    else
        v = graph_pool[b * DD + (j - 2 * DD)] / fmaxf(nn_cnt[b], 1.0f);
    pools_bf[(size_t)b * 3 * DD + j] = (__bf16)v;
}

// g_new = lrelu(pools @ Wnu + bnu)   (rows padded to 112, only <B stored)
__global__ __launch_bounds__(32) void graph_update_kernel(
    const __bf16* __restrict__ pools_bf, const __bf16* __restrict__ Wnu_t,
    const float* __restrict__ bnu, float* __restrict__ g_new, int B) {
    int lane = threadIdx.x;
    int r0 = blockIdx.x * 16;
    int mrow = lane & 15, half = lane >> 4;
    const __bf16* ar = pools_bf + (size_t)(r0 + mrow) * (3 * DD);

    v8f acc[8] = {};
#pragma unroll
    for (int ks = 0; ks < 12; ++ks) {
        v16bf a = load_a_bf16(ar, half, ks * 32);
#pragma unroll
        for (int ct = 0; ct < 8; ++ct)
            acc[ct] = wmma_bf16(a, load_b(Wnu_t, 3 * DD, ct * 16 + mrow, half, ks * 32),
                                acc[ct]);
    }
#pragma unroll
    for (int ct = 0; ct < 8; ++ct) {
        int col = ct * 16 + mrow;
        float bv = bnu[col];
#pragma unroll
        for (int g = 0; g < 8; ++g) {
            int er = r0 + g + (half << 3);
            if (er < B) g_new[(size_t)er * DD + col] = lrelu(acc[ct][g] + bv);
        }
    }
}

// out_gnode[n] = g_new[n2g[n]] + graph_feats[n]
__global__ void gnode_kernel(const float* __restrict__ g_new,
                             const int* __restrict__ n2g,
                             const float* __restrict__ graph_feats,
                             float* __restrict__ out_gnode, int N) {
    long total = (long)N * DD;
    long stride = (long)gridDim.x * blockDim.x;
    for (long i = (long)blockIdx.x * blockDim.x + threadIdx.x; i < total; i += stride) {
        int n = (int)(i >> 7), c = (int)(i & (DD - 1));
        out_gnode[i] = g_new[(size_t)n2g[n] * DD + c] + graph_feats[i];
    }
}

// ---------------------------------------------------------------------------
extern "C" void kernel_launch(void* const* d_in, const int* in_sizes, int n_in,
                              void* d_out, int out_size, void* d_ws, size_t ws_size,
                              hipStream_t stream) {
    const float* node_feats  = (const float*)d_in[0];
    const float* edge_feats  = (const float*)d_in[1];
    const float* graph_feats = (const float*)d_in[2];
    const int*   src         = (const int*)d_in[3];
    const int*   dst         = (const int*)d_in[4];
    const int*   n2g         = (const int*)d_in[5];
    const float* Wn  = (const float*)d_in[6];
    const float* bn  = (const float*)d_in[7];
    const float* We  = (const float*)d_in[8];
    const float* be  = (const float*)d_in[9];
    const float* Wg  = (const float*)d_in[10];
    const float* bg  = (const float*)d_in[11];
    const float* Weu = (const float*)d_in[12];
    const float* beu = (const float*)d_in[13];
    const float* Wnu = (const float*)d_in[14];
    const float* bnu = (const float*)d_in[15];

    const int N = in_sizes[0] / DD;  // 50000
    const int E = in_sizes[3];       // 800000
    const int B = 100;
    const int Bpad = 112;            // 7 * 16

    // output layout: node_out [N*D] | edge_out [E*D] | gnode_out [N*D]
    float* out_node  = (float*)d_out;
    float* out_edge  = out_node + (size_t)N * DD;
    float* out_gnode = out_edge + (size_t)E * DD;

    // workspace carve-out (256B aligned slabs)
    char* cur = (char*)d_ws;
    auto alloc = [&](size_t bytes) -> char* {
        char* p = cur;
        cur += (bytes + 255) & ~(size_t)255;
        return p;
    };
    __bf16* h_bf = (__bf16*)alloc((size_t)N * DD * 2);
    __bf16* u_bf = (__bf16*)alloc((size_t)N * DD * 2);
    __bf16* f_bf = (__bf16*)alloc((size_t)E * DD * 2);
    char* zstart = cur;  // everything below is atomically accumulated -> zero it
    float* hf_sum     = (float*)alloc((size_t)N * DD * 4);
    float* deg        = (float*)alloc((size_t)N * 4);
    float* node_pool  = (float*)alloc((size_t)B * DD * 4);
    float* edge_pool  = (float*)alloc((size_t)B * DD * 4);
    float* graph_pool = (float*)alloc((size_t)B * DD * 4);
    float* nn_cnt     = (float*)alloc((size_t)B * 4);
    float* ne_cnt     = (float*)alloc((size_t)B * 4);
    __bf16* pools_bf  = (__bf16*)alloc((size_t)Bpad * 3 * DD * 2);
    char* zend = cur;
    float* g_new  = (float*)alloc((size_t)B * DD * 4);
    __bf16* Wn_t  = (__bf16*)alloc((size_t)DD * DD * 2);
    __bf16* We_t  = (__bf16*)alloc((size_t)DD * DD * 2);
    __bf16* Wg_t  = (__bf16*)alloc((size_t)DD * DD * 2);
    __bf16* Weu_t = (__bf16*)alloc((size_t)4 * DD * DD * 2);
    __bf16* Wnu_t = (__bf16*)alloc((size_t)3 * DD * DD * 2);
    (void)ws_size; (void)n_in; (void)out_size;

    // 1) zero accumulation scratch
    size_t zwords = (size_t)(zend - zstart) / 4;
    zero_kernel<<<4096, 256, 0, stream>>>((uint32_t*)zstart, zwords);

    // 2) weight transpose+convert to bf16
    transpose_w_kernel<<<(DD * DD + 255) / 256, 256, 0, stream>>>(Wn, Wn_t, DD);
    transpose_w_kernel<<<(DD * DD + 255) / 256, 256, 0, stream>>>(We, We_t, DD);
    transpose_w_kernel<<<(DD * DD + 255) / 256, 256, 0, stream>>>(Wg, Wg_t, DD);
    transpose_w_kernel<<<(4 * DD * DD + 255) / 256, 256, 0, stream>>>(Weu, Weu_t, 4 * DD);
    transpose_w_kernel<<<(3 * DD * DD + 255) / 256, 256, 0, stream>>>(Wnu, Wnu_t, 3 * DD);

    // 3) degrees / per-graph counts / graph_pool
    counts_kernel<<<8192, 256, 0, stream>>>(dst, n2g, graph_feats, deg, nn_cnt, ne_cnt,
                                            graph_pool, N, E);

    // 4) projections -> bf16 activations
    int tilesN = (N + 15) / 16;
    int tilesE = (E + 15) / 16;
    proj_kernel<<<tilesN, 32, 0, stream>>>(node_feats, Wn_t, bn, h_bf, N);
    proj_kernel<<<tilesE, 32, 0, stream>>>(edge_feats, We_t, be, f_bf, E);
    proj_kernel<<<tilesN, 32, 0, stream>>>(graph_feats, Wg_t, bg, u_bf, N);

    // 5) edge update (big GEMM, gathered A) + residual + scatter-sums
    edge_update_kernel<<<tilesE, 32, 0, stream>>>(h_bf, f_bf, u_bf, Weu_t, beu, src, dst,
                                                  n2g, edge_feats, out_edge, hf_sum,
                                                  edge_pool, E);

    // 6) node update + residual + node_pool
    node_update_kernel<<<tilesN, 32, 0, stream>>>(h_bf, hf_sum, deg, u_bf, Wnu_t, bnu,
                                                  n2g, node_feats, out_node, node_pool, N);

    // 7) pooled graph state
    pool_finalize_kernel<<<(B * 3 * DD + 255) / 256, 256, 0, stream>>>(
        node_pool, edge_pool, graph_pool, nn_cnt, ne_cnt, pools_bf, B);
    graph_update_kernel<<<Bpad / 16, 32, 0, stream>>>(pools_bf, Wnu_t, bnu, g_new, B);

    // 8) scatter graph state back to nodes + residual
    gnode_kernel<<<8192, 256, 0, stream>>>(g_new, n2g, graph_feats, out_gnode, N);
}
